// Head_70626442216029
// MI455X (gfx1250) — compile-verified, matrix operand order checked by reading, not run on hardware
//
#include <hip/hip_runtime.h>

#define BATCH 4096
#define TT    33
#define CC    512
#define DKK   64
#define NOUT  192              // 3 * DKK packed (q|k|v)
#define XP    528              // LDS pitch in bf16 elems (512 + 16 pad -> no bank conflicts)

// LDS layout (dynamic shared):
//   [0, 50688)            : x as bf16, 48 rows x XP pitch
//   [50688, 76032)        : qkv fp32, 3 * 33 * 64 floats
//   [76032, 76576)        : per-wave softmax rows, 4 * 34 floats
#define LDS_X_BYTES   (48 * XP * 2)
#define LDS_QKV_FLT   (3 * TT * DKK)
#define LDS_TOTAL     (LDS_X_BYTES + (LDS_QKV_FLT + 4 * 34) * 4)

typedef __attribute__((ext_vector_type(16))) __bf16 v16bf;
typedef __attribute__((ext_vector_type(8)))  __bf16 v8bf;
typedef __attribute__((ext_vector_type(8)))  float  v8f;

// float -> bf16, round-to-nearest-even
__device__ __forceinline__ __bf16 f2bf(float f) {
  unsigned u = __builtin_bit_cast(unsigned, f);
  u += 0x7FFFu + ((u >> 16) & 1u);
  unsigned short h = (unsigned short)(u >> 16);
  return __builtin_bit_cast(__bf16, h);
}

// ---- Pre-pass: pack Wq|Wk|Wv (each [64,512] fp32) into bf16 [192,512] in workspace ----
__global__ __launch_bounds__(256)
void pack_weights_bf16(const float* __restrict__ Wq, const float* __restrict__ Wk,
                       const float* __restrict__ Wv, __bf16* __restrict__ wbf) {
  int idx = blockIdx.x * 256 + threadIdx.x;           // over NOUT*CC = 98304
  if (idx >= NOUT * CC) return;
  int row = idx >> 9;
  int col = idx & (CC - 1);
  int proj = row >> 6;
  int r = row & 63;
  const float* w = (proj == 0) ? Wq : ((proj == 1) ? Wk : Wv);
  wbf[idx] = f2bf(w[r * CC + col]);
}

// A fragment: 16x32 bf16 tile of X from LDS (ISA 16-bit A layout).
__device__ __forceinline__ v16bf load_a_frag(const __bf16* lds_x, int m16, int lane, int k0) {
  int row = m16 + (lane & 15);
  int lo  = (lane >> 4) << 3;                          // 0 or 8
  const __bf16* p = lds_x + row * XP + k0 + lo;
  v8bf l8 = *(const v8bf*)(p);                         // ds_load_b128
  v8bf h8 = *(const v8bf*)(p + 16);                    // ds_load_b128
  v16bf a;
#pragma unroll
  for (int i = 0; i < 8; ++i) { a[i] = l8[i]; a[i + 8] = h8[i]; }
  return a;
}

// B fragment: 32x16 bf16 (KxN) from packed weights; lane -> N column, same K split as A.
__device__ __forceinline__ v16bf load_b_frag(const __bf16* wbf, int d, int lane, int k0) {
  int lo = (lane >> 4) << 3;
  const __bf16* p = wbf + d * CC + k0 + lo;
  v8bf l8 = *(const v8bf*)(p);                         // global_load_b128
  v8bf h8 = *(const v8bf*)(p + 16);
  v16bf b;
#pragma unroll
  for (int i = 0; i < 8; ++i) { b[i] = l8[i]; b[i + 8] = h8[i]; }
  return b;
}

// ---- Fused: one block per batch element ----
__global__ __launch_bounds__(128)
void fused_head_kernel(const float* __restrict__ x, const __bf16* __restrict__ wbf,
                       float* __restrict__ out) {
  extern __shared__ __align__(16) unsigned char lds_raw[];
  __bf16* lds_x   = (__bf16*)lds_raw;
  float*  lds_qkv = (float*)(lds_raw + LDS_X_BYTES);
  float*  lds_p   = lds_qkv + LDS_QKV_FLT;

  const int tid  = threadIdx.x;
  const int lane = tid & 31;
  const int wave = tid >> 5;
  const int b    = blockIdx.x;
  const float* xb = x + (size_t)b * (TT * CC);

  // Stage X fp32 -> bf16 into LDS; zero-pad rows [33,48).
  for (int i = tid; i < 48 * (CC / 4); i += 128) {
    int row = i >> 7;
    int c4  = (i & 127) << 2;
    float4 v4;
    if (row < TT) v4 = *(const float4*)(xb + row * CC + c4);
    else { v4.x = 0.f; v4.y = 0.f; v4.z = 0.f; v4.w = 0.f; }
    __bf16* dst = lds_x + row * XP + c4;
    dst[0] = f2bf(v4.x); dst[1] = f2bf(v4.y); dst[2] = f2bf(v4.z); dst[3] = f2bf(v4.w);
  }
  __syncthreads();

  // Projection: Y[48 x 192] = X[48 x 512] * Wall^T.  Wave w owns N-tiles {3w, 3w+1, 3w+2}.
  // qkv lives in its own LDS region, so each tile is stored right after its K-loop:
  // only 3 accumulators (24 VGPRs) are ever live.
  for (int i = 0; i < 3; ++i) {
    const int nt = wave * 3 + i;
    const int d  = (nt << 4) + (lane & 15);            // output column 0..191
    v8f a0 = {}, a1 = {}, a2 = {};
#pragma unroll 4
    for (int ks = 0; ks < 16; ++ks) {
      const int k0 = ks << 5;
      v16bf bf = load_b_frag(wbf, d, lane, k0);
      v16bf x0 = load_a_frag(lds_x,  0, lane, k0);
      v16bf x1 = load_a_frag(lds_x, 16, lane, k0);
      v16bf x2 = load_a_frag(lds_x, 32, lane, k0);
      a0 = __builtin_amdgcn_wmma_f32_16x16x32_bf16(false, x0, false, bf, (short)0, a0, false, false);
      a1 = __builtin_amdgcn_wmma_f32_16x16x32_bf16(false, x1, false, bf, (short)0, a1, false, false);
      a2 = __builtin_amdgcn_wmma_f32_16x16x32_bf16(false, x2, false, bf, (short)0, a2, false, false);
    }
    // C/D layout: VGPR r -> (M = r + 8*laneHalf + 16*m, N = lane&15)
    const int half = lane >> 4;
    const int col  = ((nt & 3) << 4) + (lane & 15);
    float* base = lds_qkv + (nt >> 2) * (TT * DKK);
#pragma unroll
    for (int r = 0; r < 8; ++r) {
      int row0 = r + half * 8;
      base[row0 * DKK + col] = a0[r];                  // rows 0..15 always < 33
      base[(row0 + 16) * DKK + col] = a1[r];           // rows 16..31 always < 33
      int row2 = row0 + 32;
      if (row2 < TT) base[row2 * DKK + col] = a2[r];   // only row 32 survives
    }
  }
  __syncthreads();

  const float* q = lds_qkv;
  const float* k = lds_qkv + TT * DKK;
  const float* v = lds_qkv + 2 * TT * DKK;
  float* p = lds_p + wave * 34;
  const float scale = 0.04419417382415922f;            // 512^-0.5 (faithful: C, not DK)

  for (int t = wave; t < TT; t += 4) {
    const int j0 = lane, j1 = lane + 32;               // j1 valid only for lane 0 at t==32
    float s0 = -3.0e38f, s1 = -3.0e38f;
    if (j0 <= t) {
      float a = 0.f;
#pragma unroll 8
      for (int dd = 0; dd < DKK; ++dd) a += q[t * DKK + dd] * k[j0 * DKK + dd];
      s0 = a * scale;
    }
    if (j1 <= t) {
      float a = 0.f;
#pragma unroll 8
      for (int dd = 0; dd < DKK; ++dd) a += q[t * DKK + dd] * k[j1 * DKK + dd];
      s1 = a * scale;
    }
    float mx = fmaxf(s0, s1);
#pragma unroll
    for (int off = 16; off > 0; off >>= 1) mx = fmaxf(mx, __shfl_xor(mx, off, 32));
    float e0 = (j0 <= t) ? __expf(s0 - mx) : 0.f;
    float e1 = (j1 <= t) ? __expf(s1 - mx) : 0.f;
    float sum = e0 + e1;
#pragma unroll
    for (int off = 16; off > 0; off >>= 1) sum += __shfl_xor(sum, off, 32);
    float inv = 1.0f / sum;
    if (j0 <= t) p[j0] = e0 * inv;
    if (j1 <= t) p[j1] = e1 * inv;

#pragma unroll
    for (int rep = 0; rep < 2; ++rep) {
      const int dd = lane + rep * 32;
      float o = 0.f;
      for (int j = 0; j <= t; ++j) o += p[j] * v[j * DKK + dd];
      out[((size_t)b * TT + t) * DKK + dd] = o;
    }
  }
}

extern "C" void kernel_launch(void* const* d_in, const int* in_sizes, int n_in,
                              void* d_out, int out_size, void* d_ws, size_t ws_size,
                              hipStream_t stream) {
  const float* x  = (const float*)d_in[0];
  const float* Wq = (const float*)d_in[1];
  const float* Wk = (const float*)d_in[2];
  const float* Wv = (const float*)d_in[3];
  float* out = (float*)d_out;
  __bf16* wbf = (__bf16*)d_ws;                         // 192*512*2 = 196608 B of scratch

  pack_weights_bf16<<<(NOUT * CC + 255) / 256, 256, 0, stream>>>(Wq, Wk, Wv, wbf);
  fused_head_kernel<<<BATCH, 128, LDS_TOTAL, stream>>>(x, wbf, out);
}
